// MHA_28329604284693
// MI455X (gfx1250) — matrix operand order repaired and checked by reading, not compiled, not run on hardware
//
#include <hip/hip_runtime.h>

// ---------------------------------------------------------------------------
// MHA with ALiBi + causal softmax for MI455X (gfx1250), wave32, WMMA f16,
// LDS-staged GEMMs fed by the Tensor Data Mover (tensor_load_to_lds).
// B=2, S=2048, D=2048, H=16, hd=128.
// ---------------------------------------------------------------------------

typedef __attribute__((ext_vector_type(16))) _Float16 v16h;
typedef __attribute__((ext_vector_type(8)))  _Float16 v8h;
typedef __attribute__((ext_vector_type(8)))  float    v8f;
typedef __attribute__((ext_vector_type(4)))  unsigned int u32x4;
typedef __attribute__((ext_vector_type(8)))  int      i32x8;
typedef __attribute__((ext_vector_type(4)))  int      i32x4;
typedef _Float16 f16;

#define BATCH 2
#define SEQ   2048
#define DMODEL 2048
#define KDIM  2048
#define NHEAD 16
#define HDIM  128
#define N3    (3 * DMODEL)          // 6144
#define MROWS (BATCH * SEQ)         // 4096
#define LOG2E 1.4426950408889634f
#define NEG_BIG (-1.0e30f)

#if __has_builtin(__builtin_amdgcn_tensor_load_to_lds) && \
    __has_builtin(__builtin_amdgcn_s_wait_tensorcnt)
#define HAS_TDM 1
#else
#define HAS_TDM 0
#endif

// ---------------- WMMA helpers -------------------------------------------

__device__ __forceinline__ v8f wmma16(v16h a, v16h b, v8f c) {
  // D = A(16x32 f16) * B(32x16 f16) + C(16x16 f32)
  return __builtin_amdgcn_wmma_f32_16x16x32_f16(
      /*neg_a=*/false, a, /*neg_b=*/false, b,
      /*c_mod=*/(short)0, c, /*reuse_a=*/false, /*reuse_b=*/false);
}

// A-tile load, row-major A[row][k], lda in halves.
// ISA 16-bit A 16x32 layout: lanes 0-15 hold K={0..7,16..23}, lanes 16-31
// hold K={8..15,24..31} for row M=lane&15.
__device__ __forceinline__ v16h load_a16(const f16* __restrict__ A, int lda,
                                         int row, int k0, int lane) {
  const int base = (lane >> 4) * 8;
  const f16* p = A + (size_t)row * lda + k0 + base;
  v8h lo = *reinterpret_cast<const v8h*>(p);        // K = k0+base .. +7
  v8h hi = *reinterpret_cast<const v8h*>(p + 16);   // K = k0+16+base .. +7
  v16h r;
#pragma unroll
  for (int i = 0; i < 8; ++i) { r[i] = lo[i]; r[i + 8] = hi[i]; }
  return r;
}

// B-tile load from K-major storage BT[col][k], ldb in halves.
// ISA 16-bit B 32x16 layout: lanes 0-15 hold K={0..7,16..23} of column
// N=lane&15, lanes 16-31 hold K={8..15,24..31} (same order as A).
__device__ __forceinline__ v16h load_b16(const f16* __restrict__ BT, int ldb,
                                         int col, int k0, int lane) {
  const f16* p = BT + (size_t)col * ldb + k0 + ((lane >> 4) * 8);
  v8h lo = *reinterpret_cast<const v8h*>(p);
  v8h hi = *reinterpret_cast<const v8h*>(p + 16);
  v16h r;
#pragma unroll
  for (int i = 0; i < 8; ++i) { r[i] = lo[i]; r[i + 8] = hi[i]; }
  return r;
}

// 16-lane half-group reductions (wave32: C rows live across 16 lanes).
__device__ __forceinline__ float hmax16(float x) {
#pragma unroll
  for (int m = 1; m <= 8; m <<= 1) x = fmaxf(x, __shfl_xor(x, m, 32));
  return x;
}
__device__ __forceinline__ float hsum16(float x) {
#pragma unroll
  for (int m = 1; m <= 8; m <<= 1) x += __shfl_xor(x, m, 32);
  return x;
}

// ---------------- TDM: 2D tile DMA global -> LDS -------------------------
// Builds a Tensor DMA Descriptor (D#) per CDNA5 ISA ch.8:
//   group0: count=1, lds_addr, 57-bit global byte addr, type=2
//   group1: data_size=2B, tensor_dim = tile_dim (always in-bounds),
//           tile_dim0 = fast dim (elements), tile_dim1 = rows,
//           tensor_dim0_stride = row stride (elements)
//   groups 2/3 zero (2D tensor).

#if HAS_TDM
__device__ __forceinline__ void tdm_tile_load(const f16* gsrc, void* ldst,
                                              int tile_d0, int tile_d1,
                                              unsigned stride0_elems) {
  unsigned long long ga = (unsigned long long)(size_t)gsrc;
  unsigned lds_off = (unsigned)(size_t)ldst;   // addr[31:0] = LDS byte offset
  u32x4 g0;
  g0[0] = 1u;                                   // count=1, user mode
  g0[1] = lds_off;                              // lds_addr
  g0[2] = (unsigned)ga;                         // global_addr[31:0]
  g0[3] = (unsigned)((ga >> 32) & 0x01ffffffull) | 0x80000000u; // type=2
  i32x8 g1;
  g1[0] = (int)(1u << 16);                      // data_size=1 -> 2 bytes
  g1[1] = (int)(((unsigned)tile_d0 & 0xffffu) << 16);   // tensor_dim0 lo
  g1[2] = (int)(((unsigned)tile_d1 & 0xffffu) << 16);   // dim0 hi=0|dim1 lo
  g1[3] = (int)((unsigned)tile_d0 << 16);       // dim1 hi=0 | tile_dim0
  g1[4] = (int)((unsigned)tile_d1 & 0xffffu);   // tile_dim1 | tile_dim2=0
  g1[5] = (int)stride0_elems;                   // tensor_dim0_stride lo
  g1[6] = 0;                                    // stride0 hi | stride1 lo
  g1[7] = 0;                                    // stride1 hi
  i32x4 z4 = {0, 0, 0, 0};
#if defined(__clang_major__) && (__clang_major__ >= 23)
  i32x8 z8 = {0, 0, 0, 0, 0, 0, 0, 0};
  __builtin_amdgcn_tensor_load_to_lds(g0, g1, z4, z4, z8, 0);
#else
  __builtin_amdgcn_tensor_load_to_lds(g0, g1, z4, z4, 0);
#endif
}
#endif

// ---------------- conversion kernels -------------------------------------

__global__ void k_cvt_f16(const float* __restrict__ src, f16* __restrict__ dst,
                          size_t n) {
  size_t i = (size_t)blockIdx.x * blockDim.x + threadIdx.x;
  if (i < n) dst[i] = (f16)src[i];
}

// dst[c][r] = (f16)src[r][c]  (weight transposed = K-major B operand)
__global__ void k_transpose_cvt(const float* __restrict__ src,
                                f16* __restrict__ dst, int rows, int cols) {
  size_t i = (size_t)blockIdx.x * blockDim.x + threadIdx.x;
  size_t total = (size_t)rows * cols;
  if (i >= total) return;
  int r = (int)(i / cols);
  int c = (int)(i % cols);
  dst[(size_t)c * rows + r] = (f16)src[i];
}

// ---------------- LDS-staged GEMM mainloop --------------------------------
// Block: 128 threads = 4 waves.  Block tile 128(M) x 64(N), K-steps of 32.
// A staged as [128][32] halves (8KB), B as [64][32] (4KB), double buffered
// via TDM.  Each wave computes a 64x32 tile: acc[4][2] C-tiles, 8 WMMAs and
// 12 ds_load_b128 per K-step.

__device__ __forceinline__ void gemm_mainloop(
    const f16* __restrict__ Ag,   // [*][KDIM] row-major
    const f16* __restrict__ Bg,   // [*][KDIM] K-major (transposed weights)
    f16* As,                      // LDS [2][128*32]
    f16* Bs,                      // LDS [2][64*32]
    int m_blk, int n_blk, v8f acc[4][2]) {
  const int tid = threadIdx.x;
  const int lane = tid & 31;
  const int wave = tid >> 5;
  const int wm = (wave >> 1) * 64;   // wave M offset in block tile
  const int wn = (wave & 1) * 32;    // wave N offset in block tile

#if HAS_TDM
  if (tid < 32) {
    tdm_tile_load(Ag + (size_t)m_blk * KDIM, As, 32, 128, KDIM);
    tdm_tile_load(Bg + (size_t)n_blk * KDIM, Bs, 32, 64, KDIM);
  }
  int cur = 0;
  for (int k0 = 0; k0 < KDIM; k0 += 32) {
    if (tid < 32) {
      if (k0 + 32 < KDIM) {
        int nxt = cur ^ 1;
        tdm_tile_load(Ag + (size_t)m_blk * KDIM + (k0 + 32), As + nxt * 4096,
                      32, 128, KDIM);
        tdm_tile_load(Bg + (size_t)n_blk * KDIM + (k0 + 32), Bs + nxt * 2048,
                      32, 64, KDIM);
        __builtin_amdgcn_s_wait_tensorcnt(2);  // current buffer's 2 ops done
      } else {
        __builtin_amdgcn_s_wait_tensorcnt(0);
      }
    }
    __syncthreads();
    const f16* Ac = As + cur * 4096;
    const f16* Bc = Bs + cur * 2048;
    v16h a[4], b[2];
#pragma unroll
    for (int i = 0; i < 4; ++i)
      a[i] = load_a16(Ac, 32, wm + i * 16 + (lane & 15), 0, lane);
#pragma unroll
    for (int j = 0; j < 2; ++j)
      b[j] = load_b16(Bc, 32, wn + j * 16 + (lane & 15), 0, lane);
#pragma unroll
    for (int i = 0; i < 4; ++i)
#pragma unroll
      for (int j = 0; j < 2; ++j) acc[i][j] = wmma16(a[i], b[j], acc[i][j]);
    __syncthreads();   // protect buffer 'cur' before it is refilled
    cur ^= 1;
  }
#else
  // Fallback: cooperative global->LDS staging, single buffer.
  for (int k0 = 0; k0 < KDIM; k0 += 32) {
    __syncthreads();
#pragma unroll
    for (int c = 0; c < 4; ++c) {  // A: 128x32 halves = 512 x v8h chunks
      int chunk = tid + c * 128;
      int row = chunk >> 2, cc = (chunk & 3) * 8;
      *(v8h*)&As[row * 32 + cc] =
          *(const v8h*)(Ag + (size_t)(m_blk + row) * KDIM + k0 + cc);
    }
#pragma unroll
    for (int c = 0; c < 2; ++c) {  // B: 64x32 halves = 256 x v8h chunks
      int chunk = tid + c * 128;
      int row = chunk >> 2, cc = (chunk & 3) * 8;
      *(v8h*)&Bs[row * 32 + cc] =
          *(const v8h*)(Bg + (size_t)(n_blk + row) * KDIM + k0 + cc);
    }
    __syncthreads();
    v16h a[4], b[2];
#pragma unroll
    for (int i = 0; i < 4; ++i)
      a[i] = load_a16(As, 32, wm + i * 16 + (lane & 15), 0, lane);
#pragma unroll
    for (int j = 0; j < 2; ++j)
      b[j] = load_b16(Bs, 32, wn + j * 16 + (lane & 15), 0, lane);
#pragma unroll
    for (int i = 0; i < 4; ++i)
#pragma unroll
      for (int j = 0; j < 2; ++j) acc[i][j] = wmma16(a[i], b[j], acc[i][j]);
  }
#endif
}

// ---------------- GEMM1: qkv = x @ W_qkv, scatter to Q/K/Vt --------------

__global__ __launch_bounds__(128) void k_gemm_qkv(
    const f16* __restrict__ A,     // x_h   [4096][2048]
    const f16* __restrict__ BT,    // WqkvT [6144][2048]
    f16* __restrict__ Qb,          // [B*H][S][hd]
    f16* __restrict__ Kb,          // [B*H][S][hd]
    f16* __restrict__ Vt) {        // [B*H][hd][S]
  __shared__ __align__(16) f16 As[2][128 * 32];
  __shared__ __align__(16) f16 Bs[2][64 * 32];
  const int lane = threadIdx.x & 31;
  const int wave = threadIdx.x >> 5;
  const int m_blk = blockIdx.y * 128;
  const int n_blk = blockIdx.x * 64;

  v8f acc[4][2] = {};
  gemm_mainloop(A, BT, &As[0][0], &Bs[0][0], m_blk, n_blk, acc);

  const int i_off = (lane >> 4) << 3;  // +8 for upper half-wave
  const int col = lane & 15;
#pragma unroll
  for (int i = 0; i < 4; ++i)
#pragma unroll
    for (int j = 0; j < 2; ++j)
#pragma unroll
      for (int r = 0; r < 8; ++r) {
        int m = m_blk + (wave >> 1) * 64 + i * 16 + r + i_off;
        int n = n_blk + (wave & 1) * 32 + j * 16 + col;
        f16 hv = (f16)acc[i][j][r];
        int part = n >> 11;          // 0=q 1=k 2=v
        int rem = n & 2047;
        int h = rem >> 7;
        int d = rem & 127;
        int b = m >> 11;
        int s = m & 2047;
        size_t bh = (size_t)(b * NHEAD + h);
        if (part == 0)      Qb[(bh * SEQ + s) * HDIM + d] = hv;
        else if (part == 1) Kb[(bh * SEQ + s) * HDIM + d] = hv;
        else                Vt[(bh * HDIM + d) * SEQ + s] = hv;
      }
}

// ---------------- Flash attention: one wave per 16-query tile ------------

__global__ __launch_bounds__(128) void k_attn(
    const f16* __restrict__ Qb, const f16* __restrict__ Kb,
    const f16* __restrict__ Vt, f16* __restrict__ Yb) {
  __shared__ __align__(16) f16 psh[4][16 * 32];   // P bounce buffer per wave

  const int lane = threadIdx.x & 31;
  const int wave = threadIdx.x >> 5;
  const int tile = blockIdx.x * 4 + wave;          // 0..4095
  const int bh = tile >> 7;                        // S/16 = 128 tiles per bh
  const int q0 = (tile & 127) << 4;
  const int b = bh >> 4;
  const int h = bh & 15;

  const f16* Qp = Qb + (size_t)bh * (SEQ * HDIM);
  const f16* Kp = Kb + (size_t)bh * (SEQ * HDIM);
  const f16* Vp = Vt + (size_t)bh * (HDIM * SEQ);

  const float slope = exp2f(-8.0f / (float)(h + 1));
  const float scl = 0.08838834764831845f;          // 1/sqrt(128)
  const int i_off = (lane >> 4) << 3;
  const int col = lane & 15;

  v16h qa[4];                                      // Q resident, hd = 128
#pragma unroll
  for (int c = 0; c < 4; ++c)
    qa[c] = load_a16(Qp, HDIM, q0 + col, c * 32, lane);

  v8f acc[8] = {};
  float mrow[8], lrow[8];
#pragma unroll
  for (int r = 0; r < 8; ++r) { mrow[r] = NEG_BIG; lrow[r] = 0.0f; }

  for (int kb = 0; kb < q0 + 16; kb += 32) {
    v8f s0 = {}, s1 = {};
#pragma unroll
    for (int c = 0; c < 4; ++c) {
      v16h kt0 = load_b16(Kp, HDIM, kb + col, c * 32, lane);
      v16h kt1 = load_b16(Kp, HDIM, kb + 16 + col, c * 32, lane);
      s0 = wmma16(qa[c], kt0, s0);
      s1 = wmma16(qa[c], kt1, s1);
    }

    const int j0 = kb + col, j1 = j0 + 16;
#pragma unroll
    for (int r = 0; r < 8; ++r) {
      const int i = q0 + r + i_off;
      float x0 = (j0 <= i) ? s0[r] * scl + slope * (float)(j0 - i) : NEG_BIG;
      float x1 = (j1 <= i) ? s1[r] * scl + slope * (float)(j1 - i) : NEG_BIG;
      float mn = fmaxf(mrow[r], hmax16(fmaxf(x0, x1)));
      float resc = exp2f((mrow[r] - mn) * LOG2E);
      float p0 = exp2f((x0 - mn) * LOG2E);
      float p1 = exp2f((x1 - mn) * LOG2E);
      lrow[r] = lrow[r] * resc + hsum16(p0 + p1);
      mrow[r] = mn;
#pragma unroll
      for (int t = 0; t < 8; ++t) acc[t][r] *= resc;
      psh[wave][(r + i_off) * 32 + col] = (f16)p0;
      psh[wave][(r + i_off) * 32 + 16 + col] = (f16)p1;
    }

    // C-layout -> A-layout bounce through LDS (same-wave DS is in-order).
    {
      const int row = lane & 15;
      const int base = (lane >> 4) * 8;
      v8h lo = *reinterpret_cast<const v8h*>(&psh[wave][row * 32 + base]);
      v8h hi = *reinterpret_cast<const v8h*>(&psh[wave][row * 32 + 16 + base]);
      v16h pa;
#pragma unroll
      for (int i = 0; i < 8; ++i) { pa[i] = lo[i]; pa[i + 8] = hi[i]; }
#pragma unroll
      for (int t = 0; t < 8; ++t) {
        v16h vb = load_b16(Vp, SEQ, t * 16 + col, kb, lane);
        acc[t] = wmma16(pa, vb, acc[t]);
      }
    }
  }

#pragma unroll
  for (int t = 0; t < 8; ++t)
#pragma unroll
    for (int r = 0; r < 8; ++r) {
      float y = acc[t][r] / lrow[r];
      int row = b * SEQ + q0 + r + i_off;
      int c = h * HDIM + t * 16 + col;
      Yb[(size_t)row * DMODEL + c] = (f16)y;
    }
}

// ---------------- GEMM2: out = y @ W_o  (f32 output) ---------------------

__global__ __launch_bounds__(128) void k_gemm_out(
    const f16* __restrict__ A,   // Ybuf [4096][2048]
    const f16* __restrict__ BT,  // WoT  [2048][2048]
    float* __restrict__ out) {   // [4096][2048] f32
  __shared__ __align__(16) f16 As[2][128 * 32];
  __shared__ __align__(16) f16 Bs[2][64 * 32];
  const int lane = threadIdx.x & 31;
  const int wave = threadIdx.x >> 5;
  const int m_blk = blockIdx.y * 128;
  const int n_blk = blockIdx.x * 64;

  v8f acc[4][2] = {};
  gemm_mainloop(A, BT, &As[0][0], &Bs[0][0], m_blk, n_blk, acc);

  const int i_off = (lane >> 4) << 3;
  const int col = lane & 15;
#pragma unroll
  for (int i = 0; i < 4; ++i)
#pragma unroll
    for (int j = 0; j < 2; ++j)
#pragma unroll
      for (int r = 0; r < 8; ++r) {
        int m = m_blk + (wave >> 1) * 64 + i * 16 + r + i_off;
        int n = n_blk + (wave & 1) * 32 + j * 16 + col;
        out[(size_t)m * DMODEL + n] = acc[i][j][r];
      }
}

// ---------------- host-side launch ---------------------------------------

extern "C" void kernel_launch(void* const* d_in, const int* in_sizes, int n_in,
                              void* d_out, int out_size, void* d_ws,
                              size_t ws_size, hipStream_t stream) {
  (void)in_sizes; (void)n_in; (void)out_size; (void)ws_size;
  const float* x     = (const float*)d_in[0];  // [2,2048,2048]
  const float* W_qkv = (const float*)d_in[1];  // [2048,6144]
  const float* W_o   = (const float*)d_in[2];  // [2048,2048]
  float* out = (float*)d_out;

  char* ws = (char*)d_ws;
  size_t off = 0;
  f16* x_h   = (f16*)(ws + off); off += (size_t)MROWS * DMODEL * 2;   // 16 MB
  f16* WqkvT = (f16*)(ws + off); off += (size_t)N3 * DMODEL * 2;      // 24 MB
  f16* WoT   = (f16*)(ws + off); off += (size_t)DMODEL * DMODEL * 2;  //  8 MB
  f16* Qb    = (f16*)(ws + off); off += (size_t)MROWS * DMODEL * 2;   // 16 MB
  f16* Kb    = (f16*)(ws + off); off += (size_t)MROWS * DMODEL * 2;   // 16 MB
  f16* Vt    = (f16*)(ws + off); off += (size_t)MROWS * DMODEL * 2;   // 16 MB
  f16* Yb    = (f16*)(ws + off); off += (size_t)MROWS * DMODEL * 2;   // 16 MB

  // 1) precision conversion / weight transposition
  {
    size_t n = (size_t)MROWS * DMODEL;
    k_cvt_f16<<<(unsigned)((n + 255) / 256), 256, 0, stream>>>(x, x_h, n);
  }
  {
    size_t n = (size_t)DMODEL * N3;
    k_transpose_cvt<<<(unsigned)((n + 255) / 256), 256, 0, stream>>>(
        W_qkv, WqkvT, DMODEL, N3);
  }
  {
    size_t n = (size_t)DMODEL * DMODEL;
    k_transpose_cvt<<<(unsigned)((n + 255) / 256), 256, 0, stream>>>(
        W_o, WoT, DMODEL, DMODEL);
  }

  // 2) QKV projection (TDM-fed WMMA), scatter into Q/K and transposed V
  {
    dim3 grid(N3 / 64, MROWS / 128);  // 96 x 32
    k_gemm_qkv<<<grid, 128, 0, stream>>>(x_h, WqkvT, Qb, Kb, Vt);
  }

  // 3) flash attention with ALiBi + causal mask (WMMA QK^T and P.V)
  {
    int qtiles = BATCH * NHEAD * (SEQ / 16);  // 4096, 4 waves per block
    k_attn<<<qtiles / 4, 128, 0, stream>>>(Qb, Kb, Vt, Yb);
  }

  // 4) output projection (TDM-fed WMMA), f32 result
  {
    dim3 grid(DMODEL / 64, MROWS / 128);  // 32 x 32
    k_gemm_out<<<grid, 128, 0, stream>>>(Yb, WoT, out);
  }
}
